// Encoder_64046552318129
// MI455X (gfx1250) — matrix operand order repaired and checked by reading, not compiled, not run on hardware
//
#include <hip/hip_runtime.h>

typedef __attribute__((ext_vector_type(16))) __bf16 v16bf;
typedef __attribute__((ext_vector_type(8)))  float  v8f;

namespace {

constexpr int Nn  = 50000;
constexpr int Ee  = 250000;
constexpr int Dd  = 128;
constexpr int Rr  = 400;
constexpr int Ere = 20000;
constexpr int NBb = 10;
constexpr int Hh  = 8;
constexpr int DHh = 16;
constexpr int ENe = Ee + Nn;   // edges incl. self edges

__device__ __forceinline__ unsigned short f2bf(float f) {
  unsigned int u = __float_as_uint(f);
  unsigned int r = (u + 0x7FFFu + ((u >> 16) & 1u)) >> 16;  // round-to-nearest-even
  return (unsigned short)r;
}

union BF16Frag { v16bf v; unsigned short u[16]; };

__device__ __forceinline__ void atomicMaxF(float* addr, float val) {
  if (val >= 0.f) atomicMax((int*)addr, __float_as_int(val));
  else            atomicMin((unsigned int*)addr, __float_as_uint(val));
}

} // namespace

// ---------------------------------------------------------------------------
// C[M,128] = A[M,128] @ W[128,128] (+ bias) via v_wmma_f32_16x16x32_bf16.
// Wc: bf16, column-major (Wc[n*128+k]). One wave = one 16-row M tile.
// ---------------------------------------------------------------------------
__global__ void k_gemm(const float* __restrict__ A,
                       const unsigned short* __restrict__ Wc,
                       const float* __restrict__ bias,
                       float* __restrict__ C, int tiles) {
  int wave = threadIdx.x >> 5;
  int lane = threadIdx.x & 31;
  int tile = blockIdx.x * 8 + wave;
  if (tile >= tiles) return;                 // wave-uniform: EXEC stays all-1s
  int mr = lane & 15;                        // row within tile (A) / col (B,C)
  int hi = lane >> 4;                        // lane half
  int m0 = tile * 16;

  // A fragments for k-chunks 0..3 (16x32 bf16 layout, ISA 7.12.2)
  BF16Frag afrag[4];
#pragma unroll
  for (int kc = 0; kc < 4; ++kc) {
    const float* ap = A + (size_t)(m0 + mr) * Dd + kc * 32 + hi * 8;
    float4 x0 = *(const float4*)(ap + 0);
    float4 x1 = *(const float4*)(ap + 4);
    float4 y0 = *(const float4*)(ap + 16);
    float4 y1 = *(const float4*)(ap + 20);
    afrag[kc].u[0]  = f2bf(x0.x); afrag[kc].u[1]  = f2bf(x0.y);
    afrag[kc].u[2]  = f2bf(x0.z); afrag[kc].u[3]  = f2bf(x0.w);
    afrag[kc].u[4]  = f2bf(x1.x); afrag[kc].u[5]  = f2bf(x1.y);
    afrag[kc].u[6]  = f2bf(x1.z); afrag[kc].u[7]  = f2bf(x1.w);
    afrag[kc].u[8]  = f2bf(y0.x); afrag[kc].u[9]  = f2bf(y0.y);
    afrag[kc].u[10] = f2bf(y0.z); afrag[kc].u[11] = f2bf(y0.w);
    afrag[kc].u[12] = f2bf(y1.x); afrag[kc].u[13] = f2bf(y1.y);
    afrag[kc].u[14] = f2bf(y1.z); afrag[kc].u[15] = f2bf(y1.w);
  }

#pragma unroll
  for (int nt = 0; nt < 8; ++nt) {
    int n = nt * 16 + mr;
    v8f acc = {0.f, 0.f, 0.f, 0.f, 0.f, 0.f, 0.f, 0.f};
#pragma unroll
    for (int kc = 0; kc < 4; ++kc) {
      BF16Frag b;
      const unsigned short* wp = Wc + (size_t)n * Dd + kc * 32 + hi * 8;
      *(uint4*)&b.u[0] = *(const uint4*)(wp);
      *(uint4*)&b.u[8] = *(const uint4*)(wp + 16);
      acc = __builtin_amdgcn_wmma_f32_16x16x32_bf16(
          false, afrag[kc].v, false, b.v, (short)0, acc, false, false);
    }
    float bv = bias ? bias[n] : 0.f;
#pragma unroll
    for (int r8 = 0; r8 < 8; ++r8)
      C[(size_t)(m0 + hi * 8 + r8) * Dd + n] = acc[r8] + bv;
  }
}

// Convert stacked [nblk,128,128] f32 row-major weights to bf16 col-major blocks.
__global__ void k_prep(const float* __restrict__ src,
                       unsigned short* __restrict__ dst, int nblk) {
  int i = blockIdx.x * blockDim.x + threadIdx.x;
  if (i >= nblk * Dd * Dd) return;
  int b = i >> 14, r = i & 16383;
  int k = r >> 7,  n = r & 127;
  dst[(b << 14) + (n << 7) + k] = f2bf(src[i]);
}

__global__ void k_fill(float* p, float v, int n) {
  int i = blockIdx.x * blockDim.x + threadIdx.x;
  if (i < n) p[i] = v;
}

__global__ void k_relu(const float* __restrict__ in, float* __restrict__ out, int n) {
  int i = blockIdx.x * blockDim.x + threadIdx.x;
  if (i < n) { float v = in[i]; out[i] = v > 0.f ? v : 0.f; }
}

__global__ void k_relinit(const float* __restrict__ tab, const int* __restrict__ rf,
                          float* __restrict__ out) {
  int i = blockIdx.x * blockDim.x + threadIdx.x;
  if (i >= Rr * Dd) return;
  int r = i >> 7, k = i & 127;
  out[i] = tab[(size_t)rf[r] * Dd + k];
}

__global__ void k_count(const int* __restrict__ tail, float* __restrict__ freq) {
  int e = blockIdx.x * blockDim.x + threadIdx.x;
  if (e < Ee) atomicAdd(&freq[tail[e]], 1.f);
}

__global__ void k_selfacc(const float* __restrict__ er, const int* __restrict__ etype,
                          const int* __restrict__ tail, float* __restrict__ acc) {
  int i = blockIdx.x * blockDim.x + threadIdx.x;
  if (i >= Ee * Dd) return;
  int e = i >> 7, k = i & 127;
  atomicAdd(&acc[(size_t)tail[e] * Dd + k], er[(size_t)etype[e] * Dd + k]);
}

__global__ void k_selfdiv(float* __restrict__ acc, const float* __restrict__ freq) {
  int i = blockIdx.x * blockDim.x + threadIdx.x;
  if (i >= Nn * Dd) return;
  acc[i] /= freq[i >> 7];
}

// ---- relation-graph attention (segments = head) ----
__global__ void k_rel_attA(const float* __restrict__ T1, const float* __restrict__ T2,
                           const int* __restrict__ head, const int* __restrict__ tail,
                           const int* __restrict__ bins, const float* __restrict__ vec,
                           const float* __restrict__ abin,
                           float* __restrict__ abuf, float* __restrict__ amax) {
  int i = blockIdx.x * blockDim.x + threadIdx.x;
  if (i >= Ere * Hh) return;
  int e = i >> 3, h = i & 7;
  int hd = head[e], tl = tail[e];
  const float* p1 = T1 + (size_t)hd * Dd + h * DHh;
  const float* p2 = T2 + (size_t)tl * Dd + h * DHh;
  const float* vv = vec + h * DHh;
  float a = 0.f;
#pragma unroll
  for (int d = 0; d < DHh; ++d) {
    float pre = p1[d] + p2[d];
    pre = pre > 0.f ? pre : 0.2f * pre;
    a += pre * vv[d];
  }
  a += abin[bins[e] * Hh + h];
  abuf[i] = a;
  atomicMaxF(&amax[hd * Hh + h], a);
}

__global__ void k_rel_attB(float* __restrict__ abuf, const float* __restrict__ amax,
                           float* __restrict__ ssum, const int* __restrict__ head) {
  int i = blockIdx.x * blockDim.x + threadIdx.x;
  if (i >= Ere * Hh) return;
  int e = i >> 3, h = i & 7, s = head[e];
  float ev = __expf(abuf[i] - amax[s * Hh + h]);
  abuf[i] = ev;
  atomicAdd(&ssum[s * Hh + h], ev);
}

__global__ void k_rel_aggr(const float* __restrict__ abuf, const float* __restrict__ ssum,
                           const int* __restrict__ head, const int* __restrict__ tail,
                           const float* __restrict__ G, float* __restrict__ out) {
  int i = blockIdx.x * blockDim.x + threadIdx.x;
  if (i >= Ere * Dd) return;
  int e = i >> 7, k = i & 127, h = k >> 4;
  int s = head[e], t = tail[e];
  float beta = abuf[e * Hh + h] / (ssum[s * Hh + h] + 1e-16f);
  atomicAdd(&out[(size_t)s * Dd + k], beta * G[(size_t)t * Dd + k]);
}

// ---- entity-graph attention (segments = tail; self edges appended) ----
__global__ void k_ent_attA(const float* __restrict__ T1, const float* __restrict__ T2,
                           const float* __restrict__ T3, const float* __restrict__ RAe,
                           const int* __restrict__ head, const int* __restrict__ tail,
                           const int* __restrict__ etype, const float* __restrict__ vec,
                           float* __restrict__ abuf, float* __restrict__ amax) {
  int i = blockIdx.x * blockDim.x + threadIdx.x;
  if (i >= ENe * Hh) return;
  int e = i >> 3, h = i & 7;
  int tl, hd; const float* rp;
  if (e < Ee) { tl = tail[e]; hd = head[e]; rp = RAe + (size_t)etype[e] * Dd; }
  else        { int n = e - Ee; tl = n; hd = n; rp = T3 + (size_t)n * Dd; }
  const float* p1 = T1 + (size_t)tl * Dd + h * DHh;
  const float* p2 = T2 + (size_t)hd * Dd + h * DHh;
  const float* p3 = rp + h * DHh;
  const float* vv = vec + h * DHh;
  float a = 0.f;
#pragma unroll
  for (int d = 0; d < DHh; ++d) {
    float pre = p1[d] + p2[d] + p3[d];
    pre = pre > 0.f ? pre : 0.2f * pre;
    a += pre * vv[d];
  }
  abuf[i] = a;
  atomicMaxF(&amax[tl * Hh + h], a);
}

__global__ void k_ent_attB(float* __restrict__ abuf, const float* __restrict__ amax,
                           float* __restrict__ ssum, const int* __restrict__ tail) {
  int i = blockIdx.x * blockDim.x + threadIdx.x;
  if (i >= ENe * Hh) return;
  int e = i >> 3, h = i & 7;
  int tl = (e < Ee) ? tail[e] : (e - Ee);
  float ev = __expf(abuf[i] - amax[tl * Hh + h]);
  abuf[i] = ev;
  atomicAdd(&ssum[tl * Hh + h], ev);
}

__global__ void k_ent_aggr(const float* __restrict__ abuf, const float* __restrict__ ssum,
                           const int* __restrict__ head, const int* __restrict__ tail,
                           const int* __restrict__ etype,
                           const float* __restrict__ hG, const float* __restrict__ sG,
                           const float* __restrict__ rG, float* __restrict__ out) {
  int i = blockIdx.x * blockDim.x + threadIdx.x;
  if (i >= ENe * Dd) return;
  int e = i >> 7, k = i & 127, h = k >> 4;
  int tl, hd; const float* gr;
  if (e < Ee) { tl = tail[e]; hd = head[e]; gr = rG + (size_t)etype[e] * Dd; }
  else        { int n = e - Ee; tl = n; hd = n; gr = sG + (size_t)n * Dd; }
  float beta = abuf[e * Hh + h] / (ssum[tl * Hh + h] + 1e-16f);
  float g = hG[(size_t)hd * Dd + k] + gr[k];
  atomicAdd(&out[(size_t)tl * Dd + k], beta * g);
}

// ---------------------------------------------------------------------------
extern "C" void kernel_launch(void* const* d_in, const int* in_sizes, int n_in,
                              void* d_out, int out_size, void* d_ws, size_t ws_size,
                              hipStream_t stream) {
  (void)in_sizes; (void)n_in; (void)out_size; (void)ws_size;
  const float* feat     = (const float*)d_in[0];
  const float* ent_W    = (const float*)d_in[1];
  const float* ent_b    = (const float*)d_in[2];
  const float* rel_tab  = (const float*)d_in[3];
  const float* rW_att   = (const float*)d_in[4];
  const float* rb_att   = (const float*)d_in[5];
  const float* r_bin    = (const float*)d_in[6];
  const float* r_vec    = (const float*)d_in[7];
  const float* rW_aggr  = (const float*)d_in[8];
  const float* rb_aggr  = (const float*)d_in[9];
  const float* rW_res   = (const float*)d_in[10];
  const float* rb_res   = (const float*)d_in[11];
  const float* eW_att   = (const float*)d_in[12];
  const float* eb_att   = (const float*)d_in[13];
  const float* e_vec    = (const float*)d_in[14];
  const float* eW_aggr  = (const float*)d_in[15];
  const float* eb_aggr  = (const float*)d_in[16];
  const float* eW_res   = (const float*)d_in[17];
  const float* eb_res   = (const float*)d_in[18];
  const int* ent_head   = (const int*)d_in[19];
  const int* ent_tail   = (const int*)d_in[20];
  const int* ent_type   = (const int*)d_in[21];
  const int* rel_feat   = (const int*)d_in[22];
  const int* rel_head   = (const int*)d_in[23];
  const int* rel_tail   = (const int*)d_in[24];
  const int* bins       = (const int*)d_in[25];

  float* ENT_OUT = (float*)d_out;                  // [N,D]
  float* ER      = ENT_OUT + (size_t)Nn * Dd;      // [R,D] final rel emb

  const size_t ND = (size_t)Nn * Dd, RD = (size_t)Rr * Dd;
  float* f = (float*)d_ws;
  float* H0   = f; f += ND;
  float* H1   = f; f += ND;
  float* SREL = f; f += ND;
  float* T1   = f; f += ND;
  float* T2   = f; f += ND;
  float* T3   = f; f += ND;
  float* RES  = f; f += ND;
  float* ABUF = f; f += (size_t)ENe * Hh;
  float* FREQ = f; f += Nn;
  float* AMAXE= f; f += (size_t)Nn * Hh;
  float* SSUME= f; f += (size_t)Nn * Hh;
  float* RH0  = f; f += RD;
  float* RH1  = f; f += RD;
  float* RT1  = f; f += RD;
  float* RT2  = f; f += RD;
  float* RG   = f; f += RD;
  float* RRES = f; f += RD;
  float* RAe  = f; f += RD;
  float* RGe  = f; f += RD;
  float* RABUF= f; f += (size_t)Ere * Hh;
  float* AMAXR= f; f += Rr * Hh;
  float* SSUMR= f; f += Rr * Hh;
  unsigned short* WB = (unsigned short*)f;         // 21 blocks * 128*128 bf16

  const float NEG_INF = -__builtin_inff();
  auto cdiv = [](long a, long b) { return (int)((a + b - 1) / b); };
  auto prep = [&](const float* src, int blk0, int nblk) {
    k_prep<<<cdiv((long)nblk * Dd * Dd, 256), 256, 0, stream>>>(
        src, WB + (size_t)blk0 * Dd * Dd, nblk);
  };
  auto gemm = [&](const float* A, int blk, const float* bias, float* C, int M) {
    int tiles = M / 16;
    k_gemm<<<cdiv(tiles, 8), 256, 0, stream>>>(
        A, WB + (size_t)blk * Dd * Dd, bias, C, tiles);
  };
  auto fill = [&](float* p, float v, long n) {
    k_fill<<<cdiv(n, 256), 256, 0, stream>>>(p, v, (int)n);
  };

  // --- weight prep (bf16, col-major blocks) ---
  prep(ent_W,   0, 1);
  prep(rW_att,  1, 4);   // [LR,2D,D] -> blocks 1..4
  prep(rW_aggr, 5, 2);
  prep(rW_res,  7, 2);
  prep(eW_att,  9, 6);   // [LE,3D,D] -> blocks 9..14
  prep(eW_aggr, 15, 4);
  prep(eW_res,  19, 2);

  // --- initial embeddings ---
  gemm(feat, 0, ent_b, H0, Nn);
  k_relinit<<<cdiv((long)Rr * Dd, 256), 256, 0, stream>>>(rel_tab, rel_feat, RH0);

  // --- relation-graph layers ---
  const float* rin = RH0;
  for (int l = 0; l < 2; ++l) {
    float* rout = (l == 0) ? RH1 : ER;
    gemm(rin, 1 + 2 * l,     nullptr,          RT1, Rr);   // h@Wa1   (head side)
    gemm(rin, 2 + 2 * l,     rb_att + l * Dd,  RT2, Rr);   // h@Wa2+b (tail side)
    gemm(rin, 5 + l,         rb_aggr + l * Dd, RG,  Rr);   // h@Wg+bg
    gemm(rin, 7 + l,         rb_res + l * Dd,  RRES, Rr);  // residual
    fill(AMAXR, NEG_INF, (long)Rr * Hh);
    fill(SSUMR, 0.f,     (long)Rr * Hh);
    k_rel_attA<<<cdiv((long)Ere * Hh, 256), 256, 0, stream>>>(
        RT1, RT2, rel_head, rel_tail, bins, r_vec + l * Dd,
        r_bin + l * NBb * Hh, RABUF, AMAXR);
    k_rel_attB<<<cdiv((long)Ere * Hh, 256), 256, 0, stream>>>(RABUF, AMAXR, SSUMR, rel_head);
    k_rel_aggr<<<cdiv((long)Ere * Dd, 256), 256, 0, stream>>>(
        RABUF, SSUMR, rel_head, rel_tail, RG, RRES);
    k_relu<<<cdiv((long)Rr * Dd, 256), 256, 0, stream>>>(RRES, rout, Rr * Dd);
    rin = rout;
  }

  // --- in-degree + self_rel (er fixed across entity layers) ---
  fill(FREQ, 0.f, Nn);
  k_count<<<cdiv(Ee, 256), 256, 0, stream>>>(ent_tail, FREQ);
  fill(SREL, 0.f, (long)ND);
  k_selfacc<<<cdiv((long)Ee * Dd, 256), 256, 0, stream>>>(ER, ent_type, ent_tail, SREL);
  k_selfdiv<<<cdiv((long)ND, 256), 256, 0, stream>>>(SREL, FREQ);

  // --- entity-graph layers ---
  const float* hin = H0;
  for (int l = 0; l < 2; ++l) {
    float* hout = (l == 0) ? H1 : ENT_OUT;
    gemm(hin,  9 + 3 * l,     nullptr,          T1,  Nn);  // tA (gathered by tl)
    gemm(hin,  10 + 3 * l,    nullptr,          T2,  Nn);  // hA (by hd)
    gemm(SREL, 11 + 3 * l,    eb_att + l * Dd,  T3,  Nn);  // sA (+ba)
    gemm(ER,   11 + 3 * l,    eb_att + l * Dd,  RAe, Rr);  // rA (+ba)
    fill(AMAXE, NEG_INF, (long)Nn * Hh);
    fill(SSUME, 0.f,     (long)Nn * Hh);
    k_ent_attA<<<cdiv((long)ENe * Hh, 256), 256, 0, stream>>>(
        T1, T2, T3, RAe, ent_head, ent_tail, ent_type, e_vec + l * Dd, ABUF, AMAXE);
    k_ent_attB<<<cdiv((long)ENe * Hh, 256), 256, 0, stream>>>(ABUF, AMAXE, SSUME, ent_tail);
    gemm(hin,  15 + 2 * l,    nullptr,          T1,  Nn);  // hG (reuse T1)
    gemm(SREL, 16 + 2 * l,    eb_aggr + l * Dd, T2,  Nn);  // sG (+bg, reuse T2)
    gemm(ER,   16 + 2 * l,    eb_aggr + l * Dd, RGe, Rr);  // rG (+bg)
    gemm(hin,  19 + l,        eb_res + l * Dd,  RES, Nn);  // residual (aggr target)
    k_ent_aggr<<<cdiv((long)ENe * Dd, 256), 256, 0, stream>>>(
        ABUF, SSUME, ent_head, ent_tail, ent_type, T1, T2, RGe, RES);
    k_relu<<<cdiv((long)ND, 256), 256, 0, stream>>>(RES, hout, (int)ND);
    hin = hout;
  }
}
